// Model_8117488189560
// MI455X (gfx1250) — matrix operand order repaired and checked by reading, not compiled, not run on hardware
//
#include <hip/hip_runtime.h>
#include <cstdint>
#include <cstddef>

// ---------------------------------------------------------------------------
// MI455X (gfx1250) Informer forward pass.
// Heavy math: v_wmma_f32_16x16x32_f16.
// Tile staging: global_load_async_to_lds_b128, double-buffered (ASYNCcnt),
// with compile-time buffer selection (2 tiles per loop iteration).
// ---------------------------------------------------------------------------

typedef _Float16 half_t;
typedef __attribute__((ext_vector_type(16))) _Float16 v16h;
typedef __attribute__((ext_vector_type(8)))  _Float16 v8h;
typedef __attribute__((ext_vector_type(8)))  float    v8f;

#define DMODEL 512
#define NHEAD  8
#define DHEAD  64
#define BATCH  4
#define SEQL   2048
#define TOKENS (BATCH * SEQL)   // 8192 rows for every GEMM
#define COUT   7
#define PREDL  1024
#define LBLL   1024

// ---------------- WMMA / async helpers ------------------------------------

__device__ inline v8f wmma16(v16h a, v16h b, v8f c) {
  return __builtin_amdgcn_wmma_f32_16x16x32_f16(false, a, false, b,
                                                (short)0, c, false, false);
}

// Async DMA of 16 bytes per lane straight into LDS (ASYNCcnt-tracked).
__device__ inline void async_copy_b128(uint32_t lds_off, const void* g) {
  asm volatile("global_load_async_to_lds_b128 %0, %1, off"
               :: "v"(lds_off), "v"(g) : "memory");
}
__device__ inline void wait_async0() {
  asm volatile("s_wait_asynccnt 0x0" ::: "memory");
}
// Wait until only the most recent 6 async copies remain outstanding
// (async loads complete in order -> the older buffer's 6 copies are done).
__device__ inline void wait_async6() {
  asm volatile("s_wait_asynccnt 0x6" ::: "memory");
}
__device__ inline uint32_t lds_addr(const void* p) {
  // generic LDS pointer: low 32 bits are the LDS byte address
  return (uint32_t)(uintptr_t)p;
}

// A fragment: 16x32 f16 tile, row-major with leading dim `ld`.
// Lane (h=lane>>4, r=lane&15): elements 0..7 -> K=8h+e, 8..15 -> K=16+8h+e.
__device__ inline v16h load_a_frag(const half_t* base, int lane, int ld) {
  const int hh = lane >> 4, r = lane & 15;
  const half_t* p = base + r * ld + hh * 8;
  union { v16h v; v8h x[2]; } u;
  u.x[0] = *(const v8h*)(p);
  u.x[1] = *(const v8h*)(p + 16);
  return u.v;
}

// B fragment: 32x16 f16 tile stored n-major (K contiguous per lane).
__device__ inline v16h load_b_frag(const half_t* base, int lane, int ld) {
  const int hh = lane >> 4, n = lane & 15;
  const half_t* p = base + n * ld + hh * 8;
  union { v16h v; v8h x[2]; } u;
  u.x[0] = *(const v8h*)(p);
  u.x[1] = *(const v8h*)(p + 16);
  return u.v;
}

__device__ inline float gelu_exact(float x) {
  return 0.5f * x * (1.0f + erff(x * 0.70710678118654752f));
}

// ---------------- Aligned GEMM (M%128==0, N%64==0, K%128==0) ---------------
// C = act(A(f16, MxK row-major) * Wt(f16, NxK n-major) + bias)
// 256 threads = 8 waves (4m x 2n), tile 128x64, BK=64 (2 WMMA k-steps/stage).
// Double-buffered async-to-LDS staging; pipeline unrolled 2 tiles/iteration
// so the LDS buffer selection is compile-time constant.

__global__ __launch_bounds__(256) void wmma_gemm_aligned(
    const half_t* __restrict__ A, const half_t* __restrict__ Bt,
    const float* __restrict__ bias, float* __restrict__ C,
    half_t* __restrict__ C16, int M, int N, int K, int act)
{
  __shared__ half_t As[2][128 * 64];   // 2 x 16 KB
  __shared__ half_t Bs[2][64 * 64];    // 2 x  8 KB

  const int tid  = threadIdx.x;
  const int lane = tid & 31, wave = tid >> 5;
  const int wm = wave >> 1, wn = wave & 1;
  const int m0 = blockIdx.y * 128, n0 = blockIdx.x * 64;
  const int half_id = lane >> 4, l16 = lane & 15;

  // Precomputed staging addresses: 6 x 16B async copies per thread per tile.
  const half_t* aSrc[4]; uint32_t aDst0[4], aDst1[4];
#pragma unroll
  for (int j = 0; j < 4; ++j) {
    int u = j * 256 + tid;
    int row = u >> 3, c8 = (u & 7) * 8;          // 128 rows x 64 halves
    aSrc[j]  = A + (size_t)(m0 + row) * K + c8;
    aDst0[j] = lds_addr(&As[0][row * 64 + c8]);
    aDst1[j] = lds_addr(&As[1][row * 64 + c8]);
  }
  const half_t* bSrc[2]; uint32_t bDst0[2], bDst1[2];
#pragma unroll
  for (int j = 0; j < 2; ++j) {
    int u = j * 256 + tid;
    int n = u >> 3, k8 = (u & 7) * 8;            // 64 rows x 64 halves
    bSrc[j]  = Bt + (size_t)(n0 + n) * K + k8;
    bDst0[j] = lds_addr(&Bs[0][n * 64 + k8]);
    bDst1[j] = lds_addr(&Bs[1][n * 64 + k8]);
  }

  v8f acc[2][2] = {};

  auto stage = [&](const uint32_t* ad, const uint32_t* bd, int k0) {
#pragma unroll
    for (int j = 0; j < 4; ++j) async_copy_b128(ad[j], aSrc[j] + k0);
#pragma unroll
    for (int j = 0; j < 2; ++j) async_copy_b128(bd[j], bSrc[j] + k0);
  };
  auto computeTile = [&](const half_t* as, const half_t* bs) {
#pragma unroll
    for (int ks = 0; ks < 2; ++ks) {
      v16h af[2], bf[2];
#pragma unroll
      for (int ii = 0; ii < 2; ++ii)
        af[ii] = load_a_frag(as + (wm * 32 + ii * 16) * 64 + ks * 32, lane, 64);
#pragma unroll
      for (int j = 0; j < 2; ++j)
        bf[j] = load_b_frag(bs + (wn * 32 + j * 16) * 64 + ks * 32, lane, 64);
#pragma unroll
      for (int ii = 0; ii < 2; ++ii)
#pragma unroll
        for (int j = 0; j < 2; ++j)
          acc[ii][j] = wmma16(af[ii], bf[j], acc[ii][j]);
    }
  };

  const int nt = K >> 6;                          // K/64 tiles; even (K%128==0)
  stage(aDst0, bDst0, 0);
  for (int i = 0; i < nt; i += 2) {
    // ---- tile i (buffer 0); prefetch tile i+1 into buffer 1 ----
    stage(aDst1, bDst1, (i + 1) << 6);
    wait_async6();
    __syncthreads();
    computeTile(&As[0][0], &Bs[0][0]);
    __syncthreads();
    // ---- tile i+1 (buffer 1); prefetch tile i+2 into buffer 0 ----
    if (i + 2 < nt) { stage(aDst0, bDst0, (i + 2) << 6); wait_async6(); }
    else            { wait_async0(); }
    __syncthreads();
    computeTile(&As[1][0], &Bs[1][0]);
    __syncthreads();
  }

#pragma unroll
  for (int i = 0; i < 2; ++i)
#pragma unroll
    for (int j = 0; j < 2; ++j)
#pragma unroll
      for (int r = 0; r < 8; ++r) {
        int row = m0 + wm * 32 + i * 16 + half_id * 8 + r;
        int col = n0 + wn * 32 + j * 16 + l16;
        float v = acc[i][j][r] + bias[col];
        if (act == 1) v = gelu_exact(v);
        if (C)   C[(size_t)row * N + col]   = v;
        if (C16) C16[(size_t)row * N + col] = (half_t)v;
      }
}

// ---------------- Edge GEMM (guarded; used only for the N=7 projection) ----

__global__ __launch_bounds__(256) void wmma_gemm_edge(
    const half_t* __restrict__ A, const half_t* __restrict__ Bt,
    const float* __restrict__ bias, float* __restrict__ C,
    half_t* __restrict__ C16, int M, int N, int K, int act)
{
  __shared__ half_t As[128 * 32];
  __shared__ half_t Bs[64 * 32];

  const int tid  = threadIdx.x;
  const int lane = tid & 31, wave = tid >> 5;
  const int wm = wave >> 1, wn = wave & 1;
  const int m0 = blockIdx.y * 128, n0 = blockIdx.x * 64;
  const int half_id = lane >> 4, l16 = lane & 15;

  v8f acc[2][2] = {};

  for (int k0 = 0; k0 < K; k0 += 32) {
#pragma unroll
    for (int j = 0; j < 2; ++j) {
      int u = j * 256 + tid;
      int row = u >> 2, c8 = (u & 3) * 8;
      int gr = m0 + row;
      v8h val = {};
      if (gr < M) val = *(const v8h*)(A + (size_t)gr * K + k0 + c8);
      *(v8h*)&As[row * 32 + c8] = val;
    }
    {
      int n = tid >> 2, k8 = (tid & 3) * 8;
      v8h val = {};
      if (n0 + n < N) val = *(const v8h*)(Bt + (size_t)(n0 + n) * K + k0 + k8);
      *(v8h*)&Bs[n * 32 + k8] = val;
    }
    __syncthreads();

    v16h af[2], bf[2];
#pragma unroll
    for (int i = 0; i < 2; ++i)
      af[i] = load_a_frag(&As[(wm * 32 + i * 16) * 32], lane, 32);
#pragma unroll
    for (int j = 0; j < 2; ++j)
      bf[j] = load_b_frag(&Bs[(wn * 32 + j * 16) * 32], lane, 32);
#pragma unroll
    for (int i = 0; i < 2; ++i)
#pragma unroll
      for (int j = 0; j < 2; ++j)
        acc[i][j] = wmma16(af[i], bf[j], acc[i][j]);
    __syncthreads();
  }

#pragma unroll
  for (int i = 0; i < 2; ++i)
#pragma unroll
    for (int j = 0; j < 2; ++j)
#pragma unroll
      for (int r = 0; r < 8; ++r) {
        int row = m0 + wm * 32 + i * 16 + half_id * 8 + r;
        int col = n0 + wn * 32 + j * 16 + l16;
        if (row < M && col < N) {
          float v = acc[i][j][r] + bias[col];
          if (act == 1) v = gelu_exact(v);
          if (C)   C[(size_t)row * N + col]   = v;
          if (C16) C16[(size_t)row * N + col] = (half_t)v;
        }
      }
}

// ---------------- Flash attention (dense, optional causal) ----------------
// grid (Lq/128, NHEAD, BATCH), 256 threads (8 waves), 16 queries/wave.
// K chunk staged with async-to-LDS; V transposed in LDS; P round-trips
// through per-wave LDS to convert C-layout -> A-layout for P*V.

__global__ __launch_bounds__(256) void flash_attn(
    const half_t* __restrict__ Q, const half_t* __restrict__ Km,
    const half_t* __restrict__ Vm, half_t* __restrict__ O,
    int Lq, int Lk, int causal)
{
  __shared__ half_t Ks[64 * 64];        // [key][dim]
  __shared__ half_t Vs[64 * 64];        // [dim][key]
  __shared__ half_t Ps[8 * 16 * 64];    // per-wave P tile [row][key]

  const float scale = 0.125f;           // 1/sqrt(64)
  const int tid = threadIdx.x, lane = tid & 31, wave = tid >> 5;
  const int h = blockIdx.y, b = blockIdx.z;
  const int q0 = blockIdx.x * 128 + wave * 16;
  const int half_id = lane >> 4, l16 = lane & 15;
  const size_t bqo = (size_t)b * Lq * DMODEL;
  const size_t bko = (size_t)b * Lk * DMODEL;
  const int hc = h * DHEAD;

  v16h qf[2];
#pragma unroll
  for (int ks = 0; ks < 2; ++ks)
    qf[ks] = load_a_frag(Q + bqo + (size_t)q0 * DMODEL + hc + ks * 32, lane, DMODEL);

  v8f o[4] = {};
  float mrow[8], lrow[8];
#pragma unroll
  for (int r = 0; r < 8; ++r) { mrow[r] = -1e30f; lrow[r] = 0.0f; }

  for (int kc = 0; kc < Lk; kc += 64) {
    // K chunk: straight async copy (64 keys x 64 dims)
#pragma unroll
    for (int j = 0; j < 2; ++j) {
      int u = j * 256 + tid;
      int key = u >> 3, d8 = (u & 7) * 8;
      async_copy_b128(lds_addr(&Ks[key * 64 + d8]),
                      Km + bko + (size_t)(kc + key) * DMODEL + hc + d8);
    }
    // V chunk: transpose into [dim][key]
#pragma unroll
    for (int j = 0; j < 2; ++j) {
      int u = j * 256 + tid;
      int key = u >> 3, d8 = (u & 7) * 8;
      v8h v = *(const v8h*)(Vm + bko + (size_t)(kc + key) * DMODEL + hc + d8);
#pragma unroll
      for (int e = 0; e < 8; ++e) Vs[(d8 + e) * 64 + key] = v[e];
    }
    wait_async0();
    __syncthreads();

    // S = Q * K^T
    v8f s[4] = {};
#pragma unroll
    for (int t = 0; t < 4; ++t)
#pragma unroll
      for (int ks = 0; ks < 2; ++ks)
        s[t] = wmma16(qf[ks], load_b_frag(&Ks[(t * 16) * 64 + ks * 32], lane, 64), s[t]);

    // online softmax
    float mcur[8];
#pragma unroll
    for (int r = 0; r < 8; ++r) {
      int qrow = q0 + half_id * 8 + r;
      float mv = -1e30f;
#pragma unroll
      for (int t = 0; t < 4; ++t) {
        float x = s[t][r] * scale;
        int key = kc + t * 16 + l16;
        if (causal && key > qrow) x = -1e9f;
        s[t][r] = x;
        mv = fmaxf(mv, x);
      }
#pragma unroll
      for (int m = 1; m < 16; m <<= 1) mv = fmaxf(mv, __shfl_xor(mv, m, 32));
      mcur[r] = mv;
    }
#pragma unroll
    for (int r = 0; r < 8; ++r) {
      float mn = fmaxf(mrow[r], mcur[r]);
      float alpha = __expf(mrow[r] - mn);
      float ps = 0.0f;
#pragma unroll
      for (int t = 0; t < 4; ++t) {
        float p = __expf(s[t][r] - mn);
        s[t][r] = p;
        ps += p;
      }
#pragma unroll
      for (int m = 1; m < 16; m <<= 1) ps += __shfl_xor(ps, m, 32);
      lrow[r] = lrow[r] * alpha + ps;
      mrow[r] = mn;
#pragma unroll
      for (int t = 0; t < 4; ++t) o[t][r] *= alpha;
    }

    // P -> per-wave LDS, reload as A-fragments
    half_t* Pw = &Ps[wave * 16 * 64];
#pragma unroll
    for (int t = 0; t < 4; ++t)
#pragma unroll
      for (int r = 0; r < 8; ++r)
        Pw[(half_id * 8 + r) * 64 + t * 16 + l16] = (half_t)s[t][r];

    v16h pf[2];
#pragma unroll
    for (int ks = 0; ks < 2; ++ks) pf[ks] = load_a_frag(Pw + ks * 32, lane, 64);

    // O += P * V
#pragma unroll
    for (int t = 0; t < 4; ++t)
#pragma unroll
      for (int ks = 0; ks < 2; ++ks)
        o[t] = wmma16(pf[ks], load_b_frag(&Vs[(t * 16) * 64 + ks * 32], lane, 64), o[t]);

    __syncthreads();
  }

#pragma unroll
  for (int t = 0; t < 4; ++t)
#pragma unroll
    for (int r = 0; r < 8; ++r) {
      int row = q0 + half_id * 8 + r;
      float v = o[t][r] / lrow[r];
      O[bqo + (size_t)row * DMODEL + hc + t * 16 + l16] = (half_t)v;
    }
}

// ---------------- LayerNorm (wave-per-row, fused residual) ----------------

__global__ __launch_bounds__(256) void ln_kernel(
    const float* __restrict__ A, const float* __restrict__ Rsd,
    const float* __restrict__ g, const float* __restrict__ be,
    float* __restrict__ Out, half_t* __restrict__ Out16, int Mrows)
{
  const int lane = threadIdx.x & 31, wave = threadIdx.x >> 5;
  const int row = blockIdx.x * 8 + wave;
  if (row >= Mrows) return;
  const float* a = A + (size_t)row * DMODEL;
  const float* rr = Rsd ? (Rsd + (size_t)row * DMODEL) : nullptr;

  float vals[16], s = 0.0f;
#pragma unroll
  for (int j = 0; j < 16; ++j) {
    int c = lane + j * 32;
    float v = a[c] + (rr ? rr[c] : 0.0f);
    vals[j] = v;
    s += v;
  }
#pragma unroll
  for (int m = 1; m < 32; m <<= 1) s += __shfl_xor(s, m, 32);
  const float mean = s * (1.0f / DMODEL);
  float vs = 0.0f;
#pragma unroll
  for (int j = 0; j < 16; ++j) { float d = vals[j] - mean; vs += d * d; }
#pragma unroll
  for (int m = 1; m < 32; m <<= 1) vs += __shfl_xor(vs, m, 32);
  const float inv = rsqrtf(vs * (1.0f / DMODEL) + 1e-5f);
#pragma unroll
  for (int j = 0; j < 16; ++j) {
    int c = lane + j * 32;
    float v = (vals[j] - mean) * inv * g[c] + be[c];
    if (Out)   Out[(size_t)row * DMODEL + c]   = v;
    if (Out16) Out16[(size_t)row * DMODEL + c] = (half_t)v;
  }
}

// ---------------- Embedding: conv1d(wrap,k=3) + pos + time mark -----------

__global__ __launch_bounds__(256) void embed_kernel(
    const float* __restrict__ x, const float* __restrict__ xm,
    const float* __restrict__ conv, const float* __restrict__ tw,
    float* __restrict__ out, half_t* __restrict__ out16, int L)
{
  int idx = blockIdx.x * 256 + threadIdx.x;
  if (idx >= BATCH * L * DMODEL) return;
  int d = idx & (DMODEL - 1);
  int l = (idx >> 9) % L;
  int b = idx / (DMODEL * L);

  float v = 0.0f;
#pragma unroll
  for (int k = 0; k < 3; ++k) {
    int ls = l + k - 1;
    ls = (ls + L) % L;
    const float* xr = x + ((size_t)b * L + ls) * 7;
#pragma unroll
    for (int c = 0; c < 7; ++c) v += xr[c] * conv[(d * 7 + c) * 3 + k];
  }
  int d2 = d & ~1;
  float freq = __expf(-(logf(10000.0f) / (float)DMODEL) * (float)d2);
  float ang = (float)l * freq;
  v += (d & 1) ? __cosf(ang) : __sinf(ang);
  const float* xmr = xm + ((size_t)b * L + l) * 4;
#pragma unroll
  for (int t = 0; t < 4; ++t) v += xmr[t] * tw[t * DMODEL + d];

  out[idx] = v;
  out16[idx] = (half_t)v;
}

// ---------------- small utilities ----------------------------------------

// Convert f32 weight (K x N) to f16 *transposed* (N x K, n-major) so GEMM
// B-tiles are contiguous async copies.
__global__ __launch_bounds__(256) void cvt_t_kernel(const float* __restrict__ s,
                                                    half_t* __restrict__ d,
                                                    int K, int N)
{
  int i = blockIdx.x * 256 + threadIdx.x;
  if (i >= K * N) return;
  int k = i / N, n = i - k * N;
  d[(size_t)n * K + k] = (half_t)s[i];
}

__global__ __launch_bounds__(256) void slice_kernel(const float* __restrict__ p,
                                                    float* __restrict__ out)
{
  int i = blockIdx.x * 256 + threadIdx.x;
  if (i >= BATCH * PREDL * COUT) return;
  int c = i % COUT;
  int l = (i / COUT) % PREDL;
  int b = i / (COUT * PREDL);
  out[i] = p[((size_t)b * SEQL + LBLL + l) * COUT + c];
}

// ---------------- host orchestration -------------------------------------

extern "C" void kernel_launch(void* const* d_in, const int* in_sizes, int n_in,
                              void* d_out, int out_size, void* d_ws, size_t ws_size,
                              hipStream_t stream) {
  (void)in_sizes; (void)out_size; (void)ws_size;
  auto F = [&](int i) { return (const float*)d_in[i]; };
  if (n_in < 74) return;   // unexpected flattening; bail rather than misindex

  const float* x_enc      = F(0);
  const float* x_mark_enc = F(1);
  const float* x_dec      = F(2);
  const float* x_mark_dec = F(3);

  struct AttnW { const float *bk,*bo,*bq,*bv,*wk,*wo,*wq,*wv; };
  int idx = 4;
  auto rdA = [&]() {
    AttnW a;
    a.bk = F(idx++); a.bo = F(idx++); a.bq = F(idx++); a.bv = F(idx++);
    a.wk = F(idx++); a.wo = F(idx++); a.wq = F(idx++); a.wv = F(idx++);
    return a;
  };

  // jax pytree flattening order: sorted dict keys, lists in order.
  const float* dec_conv = F(idx++); const float* dec_tw = F(idx++);
  const float* d_b1 = F(idx++);     const float* d_b2 = F(idx++);
  AttnW crossA = rdA();
  const float* d_ln1b = F(idx++); const float* d_ln1g = F(idx++);
  const float* d_ln2b = F(idx++); const float* d_ln2g = F(idx++);
  const float* d_ln3b = F(idx++); const float* d_ln3g = F(idx++);
  AttnW selfA = rdA();
  const float* d_w1 = F(idx++); const float* d_w2 = F(idx++);
  const float* dec_lnb = F(idx++); const float* dec_lng = F(idx++);
  const float* enc_conv = F(idx++); const float* enc_tw = F(idx++);
  AttnW encA[2];
  const float *e_b1[2], *e_b2[2], *e_ln1b[2], *e_ln1g[2], *e_ln2b[2], *e_ln2g[2], *e_w1[2], *e_w2[2];
  for (int l = 0; l < 2; ++l) {
    encA[l] = rdA();
    e_b1[l] = F(idx++); e_b2[l] = F(idx++);
    e_ln1b[l] = F(idx++); e_ln1g[l] = F(idx++);
    e_ln2b[l] = F(idx++); e_ln2g[l] = F(idx++);
    e_w1[l] = F(idx++); e_w2[l] = F(idx++);
  }
  const float* enc_lnb = F(idx++); const float* enc_lng = F(idx++);
  const float* proj_b  = F(idx++); const float* proj_w  = F(idx++);

  // ---- workspace arena ----
  uint8_t* base = (uint8_t*)d_ws; size_t off = 0;
  auto carve = [&](size_t bytes) -> void* {
    void* p = base + off;
    off += (bytes + 255) & ~(size_t)255;
    return p;
  };
  const size_t NF = (size_t)TOKENS * DMODEL;
  float*  bufX  = (float*)carve(NF * 4);
  float*  bufZ  = (float*)carve(NF * 4);
  float*  bufP  = (float*)carve((size_t)TOKENS * COUT * 4);
  half_t* x16   = (half_t*)carve(NF * 2);
  half_t* enc16 = (half_t*)carve(NF * 2);
  half_t* q16   = (half_t*)carve(NF * 2);
  half_t* k16   = (half_t*)carve(NF * 2);
  half_t* v16b  = (half_t*)carve(NF * 2);
  half_t* c16   = (half_t*)carve(NF * 2);
  half_t* h16   = (half_t*)carve((size_t)TOKENS * 2048 * 2);
  half_t* w16   = (half_t*)carve((size_t)2048 * 512 * 2);

  auto gemm = [&](const half_t* A, const float* W, const float* bias,
                  float* C, half_t* C16, int M, int N, int K, int act) {
    size_t nw = (size_t)K * N;
    cvt_t_kernel<<<dim3((unsigned)((nw + 255) / 256)), 256, 0, stream>>>(W, w16, K, N);
    if ((N & 63) == 0 && (K & 127) == 0 && (M & 127) == 0)
      wmma_gemm_aligned<<<dim3(N / 64, M / 128), 256, 0, stream>>>(
          A, w16, bias, C, C16, M, N, K, act);
    else
      wmma_gemm_edge<<<dim3((N + 63) / 64, (M + 127) / 128), 256, 0, stream>>>(
          A, w16, bias, C, C16, M, N, K, act);
  };
  auto ln = [&](const float* a, const float* r, const float* g, const float* be,
                float* o, half_t* o16) {
    ln_kernel<<<TOKENS / 8, 256, 0, stream>>>(a, r, g, be, o, o16, TOKENS);
  };
  auto attn = [&](const AttnW& w, const half_t* qin, const half_t* kvin,
                  int causal, float* zout) {
    gemm(qin,  w.wq, w.bq, nullptr, q16,  TOKENS, DMODEL, DMODEL, 0);
    gemm(kvin, w.wk, w.bk, nullptr, k16,  TOKENS, DMODEL, DMODEL, 0);
    gemm(kvin, w.wv, w.bv, nullptr, v16b, TOKENS, DMODEL, DMODEL, 0);
    flash_attn<<<dim3(SEQL / 128, NHEAD, BATCH), 256, 0, stream>>>(
        q16, k16, v16b, c16, SEQL, SEQL, causal);
    gemm(c16, w.wo, w.bo, zout, nullptr, TOKENS, DMODEL, DMODEL, 0);
  };

  // ---- encoder ----
  embed_kernel<<<(TOKENS * DMODEL) / 256, 256, 0, stream>>>(
      x_enc, x_mark_enc, enc_conv, enc_tw, bufX, x16, SEQL);
  for (int l = 0; l < 2; ++l) {
    attn(encA[l], x16, x16, 0, bufZ);
    ln(bufX, bufZ, e_ln1g[l], e_ln1b[l], bufX, x16);
    gemm(x16, e_w1[l], e_b1[l], nullptr, h16, TOKENS, 2048, DMODEL, 1);   // GELU
    gemm(h16, e_w2[l], e_b2[l], bufZ, nullptr, TOKENS, DMODEL, 2048, 0);
    ln(bufX, bufZ, e_ln2g[l], e_ln2b[l], bufX, x16);
  }
  ln(bufX, nullptr, enc_lng, enc_lnb, bufX, enc16);

  // ---- decoder ----
  embed_kernel<<<(TOKENS * DMODEL) / 256, 256, 0, stream>>>(
      x_dec, x_mark_dec, dec_conv, dec_tw, bufX, x16, SEQL);
  attn(selfA, x16, x16, 1, bufZ);                       // causal self-attn
  ln(bufX, bufZ, d_ln1g, d_ln1b, bufX, x16);
  attn(crossA, x16, enc16, 0, bufZ);                    // cross-attn vs encoder
  ln(bufX, bufZ, d_ln2g, d_ln2b, bufX, x16);
  gemm(x16, d_w1, d_b1, nullptr, h16, TOKENS, 2048, DMODEL, 1);
  gemm(h16, d_w2, d_b2, bufZ, nullptr, TOKENS, DMODEL, 2048, 0);
  ln(bufX, bufZ, d_ln3g, d_ln3b, bufX, x16);
  ln(bufX, nullptr, dec_lng, dec_lnb, bufX, x16);

  // ---- projection + slice ----
  gemm(x16, proj_w, proj_b, bufP, nullptr, TOKENS, COUT, DMODEL, 0);
  slice_kernel<<<(BATCH * PREDL * COUT + 255) / 256, 256, 0, stream>>>(
      bufP, (float*)d_out);
}
